// EfficientAttention_68101001445756
// MI455X (gfx1250) — matrix operand order repaired and checked by reading, not compiled, not run on hardware
//
#include <hip/hip_runtime.h>

// ---------------------------------------------------------------------------
// Fused "EfficientAttention": Q/K/V projections via v_wmma_f32_16x16x32_f16,
// per-token attention (softmaxes + (softQ@sK)@V reorder) from LDS.
// Target: gfx1250 (CDNA5), wave32. 256 threads = 8 waves/WGP (2 per SIMD32).
// ---------------------------------------------------------------------------

typedef _Float16 half_t;
typedef __attribute__((ext_vector_type(16))) _Float16 v16h;
typedef __attribute__((ext_vector_type(8)))  _Float16 v8h;
typedef __attribute__((ext_vector_type(4)))  _Float16 v4h;
typedef __attribute__((ext_vector_type(8)))  float    v8f;
typedef __attribute__((ext_vector_type(4)))  float    v4f;

#define DIMS        1024
#define HEADS       16
#define HEAD_DIM    64
#define TOK_PER_BLK 16
#define XH_STRIDE   1032   // f16 elements per x-tile row (+8 pad -> LDS bank spread)
#define QS_STRIDE   1028   // f32 elements per Q/K/V row   (+4 pad)
#define SMEM_BYTES  (TOK_PER_BLK*XH_STRIDE*2 + 3*TOK_PER_BLK*QS_STRIDE*4)  // 230400 B

union FragH { v16h v; v8h h8[2]; };

// ---- Kernel 1: cast a f32 matrix to f16 (weights -> workspace) -------------
__global__ void wcast_f32_to_f16(const float* __restrict__ src,
                                 half_t* __restrict__ dst, int n4) {
  int i = blockIdx.x * blockDim.x + threadIdx.x;
  if (i < n4) {
    v4f f = ((const v4f*)src)[i];
    ((v4h*)dst)[i] = __builtin_convertvector(f, v4h);
  }
}

// ---- Kernel 2: fused projections + attention per 16-token tile -------------
__global__ __launch_bounds__(256, 1)
void attn_fused(const float* __restrict__ x,
                const half_t* __restrict__ Wqh,
                const half_t* __restrict__ Wkh,
                const half_t* __restrict__ Wvh,
                float* __restrict__ out) {
  extern __shared__ char smem[];
  half_t* Xh = (half_t*)smem;                                   // 16 x 1032 f16
  float*  Qs = (float*)(smem + TOK_PER_BLK * XH_STRIDE * 2);    // 16 x 1028 f32
  float*  Ks = Qs + TOK_PER_BLK * QS_STRIDE;
  float*  Vs = Ks + TOK_PER_BLK * QS_STRIDE;
  float*  Mt = (float*)smem;  // reuses x-tile region after phase 1 (16KB < 32KB)

  const int tid  = threadIdx.x;
  const int row0 = blockIdx.x * TOK_PER_BLK;

  // -------- Phase 0: stage x tile into LDS, f32 -> f16 --------
  {
    const int r  = tid >> 4;          // 0..15 token row
    const int c0 = (tid & 15) * 64;   // 16 threads per row
    const float* src = x + (size_t)(row0 + r) * DIMS + c0;
    half_t* dst = Xh + r * XH_STRIDE + c0;
    #pragma unroll
    for (int i = 0; i < 64; i += 4) {
      v4f f = *(const v4f*)(src + i);
      *(v4h*)(dst + i) = __builtin_convertvector(f, v4h);
    }
  }
  __syncthreads();

  // -------- Phase 1: Q/K/V = x @ W^T via WMMA (f16 in, f32 acc) --------
  // 8 waves split the 1024-wide N dimension (128 cols each, 8 N-tiles).
  {
    const int lane = tid & 31;
    const int w    = tid >> 5;     // wave id 0..7
    const int m    = lane & 15;    // A row / B column within tile
    const int hi   = lane >> 4;    // lane-half selector

    const half_t* Wmats[3] = {Wqh, Wkh, Wvh};
    float*        Omats[3] = {Qs, Ks, Vs};
    #pragma unroll
    for (int mat = 0; mat < 3; ++mat) {
      const half_t* Wm = Wmats[mat];
      float*        Om = Omats[mat];
      for (int nt = 0; nt < 8; ++nt) {
        const int n0 = w * 128 + nt * 16;
        // B fragment: lane n holds W[n0+n], K = {k0+hi*8.., k0+16+hi*8..} (row-major, no transpose)
        const half_t* wrow = Wm + (size_t)(n0 + m) * DIMS + hi * 8;
        // A fragment: lane m holds x-tile row m, K = {k0+hi*8.., k0+16+hi*8..}
        const half_t* arow = Xh + m * XH_STRIDE + hi * 8;
        v8f acc = {0.f, 0.f, 0.f, 0.f, 0.f, 0.f, 0.f, 0.f};
        #pragma unroll 8
        for (int kk = 0; kk < 32; ++kk) {
          const int k0 = kk * 32;
          FragH a, b;
          a.h8[0] = *(const v8h*)(arow + k0);
          a.h8[1] = *(const v8h*)(arow + k0 + 16);
          b.h8[0] = *(const v8h*)(wrow + k0);
          b.h8[1] = *(const v8h*)(wrow + k0 + 16);
          acc = __builtin_amdgcn_wmma_f32_16x16x32_f16(
              /*neg_a=*/false, a.v, /*neg_b=*/false, b.v,
              /*c_mod=*/(short)0, acc, /*reuse_a=*/false, /*reuse_b=*/false);
        }
        // C layout: VGPR r -> token (8*hi + r), column n0 + (lane&15)
        #pragma unroll
        for (int r = 0; r < 8; ++r)
          Om[(size_t)(8 * hi + r) * QS_STRIDE + n0 + m] = acc[r];
      }
    }
  }
  __syncthreads();

  // -------- Phase 2: per-token attention (16 threads per token) --------
  const int t   = tid >> 4;   // token 0..15
  const int sub = tid & 15;
  float* Qt = Qs + t * QS_STRIDE;
  float* Kt = Ks + t * QS_STRIDE;
  float* Vt = Vs + t * QS_STRIDE;

  // (a) sK = softmax over heads of K^T (column d: softmax over 16 heads), in place
  for (int d = sub; d < HEAD_DIM; d += 16) {
    float vals[HEADS];
    float mx = -3.4e38f;
    #pragma unroll
    for (int h = 0; h < HEADS; ++h) { vals[h] = Kt[h * HEAD_DIM + d]; mx = fmaxf(mx, vals[h]); }
    float s = 0.f;
    #pragma unroll
    for (int h = 0; h < HEADS; ++h) { vals[h] = __expf(vals[h] - mx); s += vals[h]; }
    const float inv = 1.f / s;
    #pragma unroll
    for (int h = 0; h < HEADS; ++h) Kt[h * HEAD_DIM + d] = vals[h] * inv;
  }
  // (b) softQ = softmax over head_dim of Q (row h = sub), in place
  {
    float* q = Qt + sub * HEAD_DIM;
    float mx = -3.4e38f;
    for (int d = 0; d < HEAD_DIM; ++d) mx = fmaxf(mx, q[d]);
    float s = 0.f;
    for (int d = 0; d < HEAD_DIM; ++d) { float e = __expf(q[d] - mx); q[d] = e; s += e; }
    const float inv = 1.f / s;
    for (int d = 0; d < HEAD_DIM; ++d) q[d] *= inv;
  }
  __syncthreads();

  // (c) M = softQ (16x64) @ sK (64x16); each thread owns column h' = sub
  {
    float kc[HEAD_DIM];
    #pragma unroll
    for (int d = 0; d < HEAD_DIM; ++d) kc[d] = Kt[sub * HEAD_DIM + d];  // sK[d][sub]
    float* Mcol = Mt + t * (HEADS * HEADS) + sub;
    for (int h = 0; h < HEADS; ++h) {
      float a0 = 0.f;
      #pragma unroll
      for (int d = 0; d < HEAD_DIM; ++d) a0 += Qt[h * HEAD_DIM + d] * kc[d];
      Mcol[h * HEADS] = a0;
    }
  }
  __syncthreads();

  // (d) out[h, e] = sum_h' M[h,h'] * V[h'*64 + e]; each thread owns 4 e-values
  {
    float vcol[4][HEADS];
    #pragma unroll
    for (int e4 = 0; e4 < 4; ++e4) {
      const int e = sub + e4 * 16;
      #pragma unroll
      for (int h2 = 0; h2 < HEADS; ++h2) vcol[e4][h2] = Vt[h2 * HEAD_DIM + e];
    }
    const float* Mrow = Mt + t * (HEADS * HEADS);
    float* orow = out + (size_t)(row0 + t) * DIMS;
    for (int h = 0; h < HEADS; ++h) {
      float mr[HEADS];
      #pragma unroll
      for (int h2 = 0; h2 < HEADS; ++h2) mr[h2] = Mrow[h * HEADS + h2];
      #pragma unroll
      for (int e4 = 0; e4 < 4; ++e4) {
        float acc = 0.f;
        #pragma unroll
        for (int h2 = 0; h2 < HEADS; ++h2) acc += mr[h2] * vcol[e4][h2];
        orow[h * HEAD_DIM + sub + e4 * 16] = acc;
      }
    }
  }
}

// ---------------------------------------------------------------------------
extern "C" void kernel_launch(void* const* d_in, const int* in_sizes, int n_in,
                              void* d_out, int out_size, void* d_ws, size_t ws_size,
                              hipStream_t stream) {
  (void)n_in; (void)out_size; (void)ws_size;
  const float* x  = (const float*)d_in[0];
  const float* wq = (const float*)d_in[1];
  const float* wk = (const float*)d_in[2];
  const float* wv = (const float*)d_in[3];
  float* out = (float*)d_out;

  half_t* wqh = (half_t*)d_ws;               // 3 x 2MB f16 weights in workspace
  half_t* wkh = wqh + (size_t)DIMS * DIMS;
  half_t* wvh = wkh + (size_t)DIMS * DIMS;

  const int n4 = (DIMS * DIMS) / 4;          // 262144 float4 per weight matrix
  wcast_f32_to_f16<<<dim3(n4 / 256), dim3(256), 0, stream>>>(wq, wqh, n4);
  wcast_f32_to_f16<<<dim3(n4 / 256), dim3(256), 0, stream>>>(wk, wkh, n4);
  wcast_f32_to_f16<<<dim3(n4 / 256), dim3(256), 0, stream>>>(wv, wvh, n4);

  // Opt in to 225KB dynamic LDS (WGP supports up to 320KB per workgroup).
  (void)hipFuncSetAttribute((const void*)attn_fused,
                            hipFuncAttributeMaxDynamicSharedMemorySize, SMEM_BYTES);

  const int ntok = in_sizes[0] / DIMS;       // 4*4096 = 16384 tokens
  attn_fused<<<dim3(ntok / TOK_PER_BLK), dim3(256), SMEM_BYTES, stream>>>(
      x, wqh, wkh, wvh, out);
}
